// GroupDCNv4Layer_32152125178040
// MI455X (gfx1250) — matrix operand order repaired and checked by reading, not compiled, not run on hardware
//
#include <hip/hip_runtime.h>
#include <hip/hip_bf16.h>
#include <math.h>

// ---------------- problem constants ----------------
#define NB   8
#define CC   128
#define HH   96
#define WW   96
#define HW   (HH*WW)          // 9216
#define GG   8
#define GC   16               // CC/GG
#define KK   3
#define K2   9
#define OM   216              // G*K2*3
#define OMP  224              // padded to 14 WMMA tiles of 16
#define APAD 136              // A tile row stride in f16 (272B = 17*16B)
#define GN_EPS 1e-5f

typedef _Float16 v16h __attribute__((ext_vector_type(16)));
typedef _Float16 v8h  __attribute__((ext_vector_type(8)));
typedef float    v8f  __attribute__((ext_vector_type(8)));

// ---------------- K0: zero per-sample stats ----------------
__global__ void k0_zero_stats(float* stats) {
    int t = threadIdx.x;
    if (t < 32) stats[t] = 0.0f;
}

// ---------------- K1: depthwise 3x3 conv + bias, accumulate sum/sumsq per n ----------------
__global__ void k1_dwconv(const float* __restrict__ xin,
                          const float* __restrict__ dw_w,
                          const float* __restrict__ dw_b,
                          float* __restrict__ x1,
                          float* __restrict__ stats) {
    const long long idx = (long long)blockIdx.x * blockDim.x + threadIdx.x;
    const int n   = (int)(idx / ((long long)CC * HW));
    const int r   = (int)(idx % ((long long)CC * HW));
    const int c   = r / HW;
    const int pix = r % HW;
    const int y   = pix / WW;
    const int x   = pix % WW;

    const float* src = xin + ((long long)(n * CC + c)) * HW;
    const float* wv  = dw_w + c * 9;

    float acc = dw_b[c];
#pragma unroll
    for (int ky = 0; ky < 3; ++ky) {
        const int yy = y + ky - 1;
        if (yy < 0 || yy >= HH) continue;
#pragma unroll
        for (int kx = 0; kx < 3; ++kx) {
            const int xx = x + kx - 1;
            if (xx < 0 || xx >= WW) continue;
            acc += src[yy * WW + xx] * wv[ky * 3 + kx];
        }
    }
    x1[idx] = acc;

    // block reduction of sum / sumsq (one block lies entirely inside one n:
    // per-n extent = 128*9216 = 1179648, multiple of 256)
    __shared__ float rs[256];
    __shared__ float rq[256];
    rs[threadIdx.x] = acc;
    rq[threadIdx.x] = acc * acc;
    __syncthreads();
    for (int s = 128; s > 0; s >>= 1) {
        if (threadIdx.x < s) {
            rs[threadIdx.x] += rs[threadIdx.x + s];
            rq[threadIdx.x] += rq[threadIdx.x + s];
        }
        __syncthreads();
    }
    if (threadIdx.x == 0) {
        atomicAdd(&stats[n],     rs[0]);
        atomicAdd(&stats[8 + n], rq[0]);
    }
}

// ---------------- K2: finalize mean/rstd; pack om_w -> f16 padded [OMP][CC]; pad om_b ----------------
__global__ void k2_prep(const float* __restrict__ om_w,
                        const float* __restrict__ om_b,
                        float* __restrict__ stats,
                        float* __restrict__ ob_pad,
                        _Float16* __restrict__ Bh) {
    const int t = threadIdx.x;
    if (t < 8) {
        const float cnt  = (float)((long long)CC * HW);
        const float mean = stats[t] / cnt;
        const float var  = stats[8 + t] / cnt - mean * mean;
        stats[16 + t] = mean;
        stats[24 + t] = __frsqrt_rn(var + GN_EPS);
    }
    for (int i = t; i < OMP * CC; i += blockDim.x) {
        const int o = i / CC, c = i % CC;
        const float v = (o < OM) ? om_w[o * CC + c] : 0.0f;
        Bh[i] = (_Float16)v;
    }
    for (int i = t; i < OMP; i += blockDim.x)
        ob_pad[i] = (i < OM) ? om_b[i] : 0.0f;
}

// ---------------- K3: fused norm+GELU -> WMMA GEMM -> deformable sampling ----------------
// block: 256 threads (8 wave32), 32 pixels. 2 M-tiles x 14 O-tiles = 28 WMMA tiles.
__global__ void __launch_bounds__(256)
k3_gemm_sample(const float* __restrict__ xin,
               const float* __restrict__ x1,
               const float* __restrict__ gn_w,
               const float* __restrict__ gn_b,
               const float* __restrict__ stats,
               const float* __restrict__ ob_pad,
               const _Float16* __restrict__ Bh,
               float* __restrict__ out) {
    __shared__ _Float16 Ald[32][APAD];    // A tile (pixels x channels), f16, padded rows
    __shared__ float    OMld[32][OMP + 8];// GEMM result (om) per pixel
    __shared__ float    BiasLd[OMP];

    const int tid    = threadIdx.x;
    const int lane   = tid & 31;
    const int waveId = tid >> 5;
    const int n      = blockIdx.x / (HW / 32);
    const int pix0   = (blockIdx.x % (HW / 32)) * 32;

    // warm B in L2 / WGP$ (global_prefetch_b8)
    __builtin_prefetch(Bh + tid * 112, 0, 1);

    const float mean = stats[16 + n];
    const float rstd = stats[24 + n];

    if (tid < OMP) BiasLd[tid] = ob_pad[tid];

    // ---- stage A: load x1 (coalesced: 32 consecutive pixels per wave),
    //      apply norm + affine + exact GELU, convert to f16 ----
    const float inv_sqrt2 = 0.70710678118654752f;
#pragma unroll
    for (int j = 0; j < 16; ++j) {
        const int i = tid + j * 256;           // 32*128 = 4096 elements
        const int c = i >> 5;                  // channel 0..127
        const int p = i & 31;                  // pixel within tile
        float v = x1[((long long)(n * CC + c)) * HW + pix0 + p];
        v = (v - mean) * rstd * gn_w[c] + gn_b[c];
        v = 0.5f * v * (1.0f + erff(v * inv_sqrt2));   // exact GELU
        Ald[p][c] = (_Float16)v;
    }
    __syncthreads();

    // ---- stage B: WMMA GEMM, D = A(32x128) x B^T(128x224) + bias -> OMld ----
    const int rsel = (lane >> 4) & 1;
    const int rlo  = lane & 15;
    for (int t = waveId; t < 28; t += 8) {
        const int mt = t / 14;                 // 0..1
        const int ot = t % 14;                 // 0..13
        v8f acc = {};
#pragma unroll
        for (int kk = 0; kk < 4; ++kk) {
            // A fragment (ISA 16-bit A 16x32): lane -> row M = rlo,
            // K chunks {sel*8..+7} and {16+sel*8..+7} within this 32-K step
            const int ka0 = kk * 32 + rsel * 8;
            const v8h alo = *(const v8h*)&Ald[mt * 16 + rlo][ka0];
            const v8h ahi = *(const v8h*)&Ald[mt * 16 + rlo][ka0 + 16];
            // B fragment (ISA B layout): lane -> column N = rlo,
            // K chunks {sel*8..+7} and {16+sel*8..+7} within this 32-K step
            const _Float16* bp = Bh + (ot * 16 + rlo) * 128 + kk * 32 + rsel * 8;
            const v8h blo = *(const v8h*)(bp);
            const v8h bhi = *(const v8h*)(bp + 16);
            v16h a, b;
#pragma unroll
            for (int q = 0; q < 8; ++q) {
                a[q] = alo[q]; a[q + 8] = ahi[q];
                b[q] = blo[q]; b[q + 8] = bhi[q];
            }
            acc = __builtin_amdgcn_wmma_f32_16x16x32_f16(
                      false, a, false, b, (short)0, acc, false, false);
        }
        // D layout: vgpr i -> M = i + (lane>=16 ? 8 : 0), N = lane&15
        const int obase = ot * 16 + rlo;
        const int madd  = (lane >> 4) << 3;
#pragma unroll
        for (int i = 0; i < 8; ++i) {
            const int m = mt * 16 + i + madd;
            OMld[m][obase] = acc[i] + BiasLd[obase];
        }
    }
    __syncthreads();

    // ---- stage C: deformable bilinear sampling; one (pixel, group) per thread ----
    const int p  = tid >> 3;               // 0..31
    const int g  = tid & 7;                // 0..7
    const int pix = pix0 + p;
    const int y0p = pix / WW;
    const int x0p = pix % WW;

    const float* xb = xin + ((long long)(n * CC + g * GC)) * HW;
    float acc[GC];
#pragma unroll
    for (int cc = 0; cc < GC; ++cc) acc[cc] = 0.0f;

    const float* omrow = &OMld[p][g * 27];
#pragma unroll
    for (int k = 0; k < K2; ++k) {
        const int ky = k / 3, kx = k % 3;
        const float offx = omrow[2 * k];
        const float offy = omrow[2 * k + 1];
        const float msk  = omrow[18 + k];
        const float ly_f = (float)(y0p + ky - 1) + offy;
        const float lx_f = (float)(x0p + kx - 1) + offx;
        const float fy = floorf(ly_f);
        const float fx = floorf(lx_f);
        const float dy = ly_f - fy;
        const float dx = lx_f - fx;
        const int yi = (int)fy;
        const int xi = (int)fx;
        const float w00 = msk * (1.0f - dy) * (1.0f - dx);
        const float w01 = msk * (1.0f - dy) * dx;
        const float w10 = msk * dy * (1.0f - dx);
        const float w11 = msk * dy * dx;
#pragma unroll
        for (int corner = 0; corner < 4; ++corner) {
            const int cy = yi + (corner >> 1);
            const int cx = xi + (corner & 1);
            const float wgt = (corner == 0) ? w00 : (corner == 1) ? w01
                             : (corner == 2) ? w10 : w11;
            if (cy >= 0 && cy < HH && cx >= 0 && cx < WW) {
                const float* src = xb + cy * WW + cx;
#pragma unroll
                for (int cc = 0; cc < GC; ++cc)
                    acc[cc] += wgt * src[(long long)cc * HW];
            }
        }
    }
    float* dst = out + ((long long)(n * CC + g * GC)) * HW + pix;
#pragma unroll
    for (int cc = 0; cc < GC; ++cc)
        dst[(long long)cc * HW] = acc[cc];
}

// ---------------- launcher ----------------
extern "C" void kernel_launch(void* const* d_in, const int* in_sizes, int n_in,
                              void* d_out, int out_size, void* d_ws, size_t ws_size,
                              hipStream_t stream) {
    const float* xin  = (const float*)d_in[0];
    const float* dw_w = (const float*)d_in[1];
    const float* dw_b = (const float*)d_in[2];
    const float* gn_w = (const float*)d_in[3];
    const float* gn_b = (const float*)d_in[4];
    const float* om_w = (const float*)d_in[5];
    const float* om_b = (const float*)d_in[6];
    float* out = (float*)d_out;

    // workspace layout (all 16B aligned)
    float*     x1     = (float*)d_ws;                        // 9437184 f32
    float*     stats  = x1 + (long long)NB * CC * HW;        // 32 f32
    float*     ob_pad = stats + 32;                          // 224 f32
    _Float16*  Bh     = (_Float16*)(ob_pad + OMP);           // 224*128 f16

    k0_zero_stats<<<1, 32, 0, stream>>>(stats);

    const long long total = (long long)NB * CC * HW;         // 9437184
    k1_dwconv<<<(unsigned)(total / 256), 256, 0, stream>>>(xin, dw_w, dw_b, x1, stats);

    k2_prep<<<1, 256, 0, stream>>>(om_w, om_b, stats, ob_pad, Bh);

    const unsigned blocks = (unsigned)((long long)NB * HW / 32);  // 2304
    k3_gemm_sample<<<blocks, 256, 0, stream>>>(xin, x1, gn_w, gn_b, stats,
                                               ob_pad, Bh, out);
}
